// decoder_RDDNet_stage5cat_decoderfinal_attention_77979426226694
// MI455X (gfx1250) — compile-verified, matrix operand order checked by reading, not run on hardware
//
#include <hip/hip_runtime.h>
#include <math.h>

typedef __attribute__((ext_vector_type(16))) _Float16     v16h;
typedef __attribute__((ext_vector_type(8)))  float        v8f;
typedef __attribute__((ext_vector_type(4)))  unsigned int u32x4;
typedef __attribute__((ext_vector_type(8)))  unsigned int u32x8;

#define EPSBN 1e-5f

__device__ __forceinline__ float bn_scale(float g, float v) { return g * rsqrtf(v + EPSBN); }

// ---------------------------------------------------------------------------
// 1x1 conv + BN + ReLU, NCHW f32 -> NCHW f32 (stems)
// ---------------------------------------------------------------------------
__global__ void k_conv1x1_bn_relu_nchw(const float* __restrict__ in, const float* __restrict__ w,
                                       const float* __restrict__ g, const float* __restrict__ be,
                                       const float* __restrict__ mn, const float* __restrict__ vr,
                                       float* __restrict__ out, int B, int Cin, int Cout, int HW) {
  int idx = blockIdx.x * blockDim.x + threadIdx.x;
  int total = B * Cout * HW;
  if (idx >= total) return;
  int pos = idx % HW;
  int t = idx / HW;
  int co = t % Cout;
  int b = t / Cout;
  const float* ip = in + (size_t)b * Cin * HW + pos;
  const float* wp = w + (size_t)co * Cin;
  float acc = 0.f;
  for (int ci = 0; ci < Cin; ++ci) acc = fmaf(ip[(size_t)ci * HW], wp[ci], acc);
  float s = bn_scale(g[co], vr[co]);
  float y = acc * s + (be[co] - mn[co] * s);
  out[idx] = fmaxf(y, 0.f);
}

// ---------------------------------------------------------------------------
// 1x1 conv + BN + ReLU, NCHW f32 -> NCHW (channel-major) f16   (V projection)
// ---------------------------------------------------------------------------
__global__ void k_conv1x1_bn_relu_nchw_h(const float* __restrict__ in, const float* __restrict__ w,
                                         const float* __restrict__ g, const float* __restrict__ be,
                                         const float* __restrict__ mn, const float* __restrict__ vr,
                                         _Float16* __restrict__ out, int B, int Cin, int Cout, int HW) {
  int idx = blockIdx.x * blockDim.x + threadIdx.x;
  int total = B * Cout * HW;
  if (idx >= total) return;
  int pos = idx % HW;
  int t = idx / HW;
  int co = t % Cout;
  int b = t / Cout;
  const float* ip = in + (size_t)b * Cin * HW + pos;
  const float* wp = w + (size_t)co * Cin;
  float acc = 0.f;
  for (int ci = 0; ci < Cin; ++ci) acc = fmaf(ip[(size_t)ci * HW], wp[ci], acc);
  float s = bn_scale(g[co], vr[co]);
  float y = acc * s + (be[co] - mn[co] * s);
  out[idx] = (_Float16)fmaxf(y, 0.f);
}

// ---------------------------------------------------------------------------
// 1x1 conv + BN + ReLU, NCHW f32 -> token-major [B,N,C] f16 (q/k)
// ---------------------------------------------------------------------------
__global__ void k_conv1x1_bn_relu_tok_h(const float* __restrict__ in, const float* __restrict__ w,
                                        const float* __restrict__ g, const float* __restrict__ be,
                                        const float* __restrict__ mn, const float* __restrict__ vr,
                                        _Float16* __restrict__ out, int B, int Cin, int Cout, int HW) {
  int idx = blockIdx.x * blockDim.x + threadIdx.x;
  int total = B * HW * Cout;
  if (idx >= total) return;
  int co = idx % Cout;
  int t = idx / Cout;
  int n = t % HW;
  int b = t / HW;
  const float* ip = in + (size_t)b * Cin * HW + n;
  const float* wp = w + (size_t)co * Cin;
  float acc = 0.f;
  for (int ci = 0; ci < Cin; ++ci) acc = fmaf(ip[(size_t)ci * HW], wp[ci], acc);
  float s = bn_scale(g[co], vr[co]);
  float y = acc * s + (be[co] - mn[co] * s);
  out[idx] = (_Float16)fmaxf(y, 0.f);
}

// ---------------------------------------------------------------------------
// bilinear resize, half-pixel centers (align_corners=False)
// ---------------------------------------------------------------------------
__global__ void k_resize_bilinear(const float* __restrict__ in, float* __restrict__ out,
                                  int B, int C, int Hi, int Wi, int Ho, int Wo) {
  int idx = blockIdx.x * blockDim.x + threadIdx.x;
  int total = B * C * Ho * Wo;
  if (idx >= total) return;
  int x = idx % Wo;
  int t = idx / Wo;
  int y = t % Ho;
  int bc = t / Ho;
  float sy = (y + 0.5f) * (float)Hi / (float)Ho - 0.5f;
  float sx = (x + 0.5f) * (float)Wi / (float)Wo - 0.5f;
  sy = fminf(fmaxf(sy, 0.f), (float)(Hi - 1));
  sx = fminf(fmaxf(sx, 0.f), (float)(Wi - 1));
  int y0 = (int)sy, x0 = (int)sx;
  int y1 = min(y0 + 1, Hi - 1), x1 = min(x0 + 1, Wi - 1);
  float fy = sy - (float)y0, fx = sx - (float)x0;
  const float* p = in + (size_t)bc * Hi * Wi;
  float v00 = p[y0 * Wi + x0], v01 = p[y0 * Wi + x1];
  float v10 = p[y1 * Wi + x0], v11 = p[y1 * Wi + x1];
  out[idx] = (v00 * (1.f - fx) + v01 * fx) * (1.f - fy) + (v10 * (1.f - fx) + v11 * fx) * fy;
}

// ---------------------------------------------------------------------------
// depthwise 3x3 conv over (inA + inB), + BN + ReLU
// ---------------------------------------------------------------------------
__global__ void k_dw3x3_add_bn_relu(const float* __restrict__ inA, const float* __restrict__ inB,
                                    const float* __restrict__ w,
                                    const float* __restrict__ g, const float* __restrict__ be,
                                    const float* __restrict__ mn, const float* __restrict__ vr,
                                    float* __restrict__ out, int B, int C, int H, int W) {
  int idx = blockIdx.x * blockDim.x + threadIdx.x;
  int total = B * C * H * W;
  if (idx >= total) return;
  int x = idx % W;
  int t = idx / W;
  int y = t % H;
  t /= H;
  int c = t % C;
  int b = t / C;
  size_t base = (size_t)(b * C + c) * H * W;
  const float* pa = inA + base;
  const float* pb = inB + base;
  const float* wp = w + c * 9;
  float acc = 0.f;
  for (int ky = 0; ky < 3; ++ky) {
    int yy = y + ky - 1;
    if (yy < 0 || yy >= H) continue;
    for (int kx = 0; kx < 3; ++kx) {
      int xx = x + kx - 1;
      if (xx < 0 || xx >= W) continue;
      int o2 = yy * W + xx;
      acc = fmaf(pa[o2] + pb[o2], wp[ky * 3 + kx], acc);
    }
  }
  float s = bn_scale(g[c], vr[c]);
  float yv = acc * s + (be[c] - mn[c] * s);
  out[idx] = fmaxf(yv, 0.f);
}

// ---------------------------------------------------------------------------
// TDM: issue a 2-D tensor tile DMA (global -> LDS), per CDNA5 D# layout.
//   Tile is tile1 rows x tile0 elements (2-byte data), row stride = stride0
//   elements; deposited contiguously in LDS at lds_off.
// ---------------------------------------------------------------------------
__device__ __forceinline__ void tdm_load_2d(unsigned lds_off, const void* gaddr,
                                            unsigned dim0, unsigned dim1,
                                            unsigned tile0, unsigned tile1,
                                            unsigned long long stride0) {
  union { unsigned w[4]; u32x4 v; } g0;
  union { unsigned w[8]; u32x8 v; } g1;
  unsigned long long ga = (unsigned long long)(uintptr_t)gaddr;
  // D# group 0: count=1 | lds_addr | global_addr[56:0] | type=2
  g0.w[0] = 1u;
  g0.w[1] = lds_off;
  g0.w[2] = (unsigned)ga;
  g0.w[3] = ((unsigned)(ga >> 32) & 0x1FFFFFFu) | (2u << 30);
  // D# group 1: data_size=2B, dims/tiles/stride
  g1.w[0] = 1u << 16;                                   // data_size code 1 (2B)
  g1.w[1] = (dim0 & 0xFFFFu) << 16;                     // tensor_dim0[15:0]
  g1.w[2] = (dim0 >> 16) | ((dim1 & 0xFFFFu) << 16);    // dim0 hi | dim1 lo
  g1.w[3] = (dim1 >> 16) | ((tile0 & 0xFFFFu) << 16);   // dim1 hi | tile_dim0
  g1.w[4] = tile1 & 0xFFFFu;                            // tile_dim1 (tile_dim2=0)
  g1.w[5] = (unsigned)stride0;                          // tensor_dim0_stride lo
  g1.w[6] = (unsigned)(stride0 >> 32) & 0xFFFFu;        // stride hi (dim1_stride=0)
  g1.w[7] = 0;
  asm volatile("tensor_load_to_lds %0, %1" :: "s"(g0.v), "s"(g1.v) : "memory");
}

// ---------------------------------------------------------------------------
// Flash attention, wave32 WMMA f32_16x16x32_f16 + TDM tile staging.
// Q/K: [B,N,128] f16 token-major. V: [B,128,N] f16 channel-major.
// ctx out: [B,128,N] f32 (NCHW).
// Block = 256 threads (8 waves); each wave owns 16 Q rows (BM=128/block).
// K tile [64][128] and V tile [128][64] DMA'd into LDS by the TDM.
// ---------------------------------------------------------------------------
#define AC  128
#define ABM 128
#define ABN 64

// 16-bit A/B fragment (16x32 / 32x16) from a per-lane row pointer:
//   low-half lanes hold K = kbase+{0..7, 16..23}, high-half K = kbase+{8..15, 24..31}
__device__ __forceinline__ v16h frag_row16(const _Float16* __restrict__ rowp, int kbase, int hi8) {
  const _Float16* p = rowp + kbase + hi8;
  v16h r;
#pragma unroll
  for (int i = 0; i < 8; ++i) { r[i] = p[i]; r[8 + i] = p[16 + i]; }
  return r;
}

__global__ __launch_bounds__(256) void k_flash_attn(const _Float16* __restrict__ qh,
                                                    const _Float16* __restrict__ kh,
                                                    const _Float16* __restrict__ vh,
                                                    float* __restrict__ ctx, int N) {
  __shared__ _Float16 ldsK[ABN * AC];        // K tile, [j][c]
  __shared__ _Float16 ldsV[AC * ABN];        // V tile, [c][j] (TDM strided tile)
  __shared__ _Float16 ldsP[8][16 * ABN];     // per-wave P scratch, [m][j]

  const int tid = threadIdx.x;
  const int wave = tid >> 5;
  const int lane = tid & 31;
  const int l16 = lane & 15;
  const int hi8 = (lane & 16) ? 8 : 0;
  const int b = blockIdx.y;
  const int qrow0 = blockIdx.x * ABM + wave * 16;

  const unsigned ldsK_off = (unsigned)(uintptr_t)(void*)ldsK;
  const unsigned ldsV_off = (unsigned)(uintptr_t)(void*)ldsV;

  // Q fragments for this wave's 16 rows (4 K-slices of 32)
  const _Float16* qrow = qh + ((size_t)b * N + qrow0 + l16) * AC;
  v16h qf[4];
#pragma unroll
  for (int f = 0; f < 4; ++f) qf[f] = frag_row16(qrow, 32 * f, hi8);

  v8f o[8];
#pragma unroll
  for (int t = 0; t < 8; ++t)
#pragma unroll
    for (int i = 0; i < 8; ++i) o[t][i] = 0.f;
  float m_i[8], l_i[8];
#pragma unroll
  for (int i = 0; i < 8; ++i) { m_i[i] = -__builtin_inff(); l_i[i] = 0.f; }

  const _Float16* kb = kh + (size_t)b * N * AC;   // [N][128]
  const _Float16* vb = vh + (size_t)b * AC * N;   // [128][N]
  const int nblk = N / ABN;

  for (int jb = 0; jb < nblk; ++jb) {
    __syncthreads();  // prior tile reads complete before TDM overwrites LDS
    if (wave == 0) {
      // K tile: 64 rows x 128 halves, contiguous rows (stride 128)
      tdm_load_2d(ldsK_off, kb + (size_t)jb * ABN * AC, AC, (unsigned)N, AC, ABN, AC);
      // V tile: 128 rows x 64 halves, row stride N (channel-major source)
      tdm_load_2d(ldsV_off, vb + (size_t)jb * ABN, (unsigned)N, AC, ABN, AC,
                  (unsigned long long)N);
      __builtin_amdgcn_s_wait_tensorcnt(0);
    }
    if (jb + 1 < nblk)  // hint next K block toward L2
      __builtin_prefetch(kb + (size_t)(jb + 1) * ABN * AC, 0, 1);
    __syncthreads();

    // S = Q * K^T : 4 column tiles x 4 k-steps
    v8f st[4];
#pragma unroll
    for (int t = 0; t < 4; ++t) {
      v8f s;
#pragma unroll
      for (int i = 0; i < 8; ++i) s[i] = 0.f;
      const _Float16* krow = ldsK + (16 * t + l16) * AC;
#pragma unroll
      for (int k = 0; k < 4; ++k)
        s = __builtin_amdgcn_wmma_f32_16x16x32_f16(false, qf[k], false,
                                                   frag_row16(krow, 32 * k, hi8),
                                                   (short)0, s, false, false);
      st[t] = s;
    }

    // online softmax: row max / scale / exp / row sum (rows = i + hi8)
    float mx[8];
#pragma unroll
    for (int i = 0; i < 8; ++i)
      mx[i] = fmaxf(fmaxf(st[0][i], st[1][i]), fmaxf(st[2][i], st[3][i]));
#pragma unroll
    for (int off = 8; off >= 1; off >>= 1)
#pragma unroll
      for (int i = 0; i < 8; ++i) mx[i] = fmaxf(mx[i], __shfl_xor(mx[i], off, 32));

    float alpha[8], rs[8];
#pragma unroll
    for (int i = 0; i < 8; ++i) {
      float mnew = fmaxf(m_i[i], mx[i]);
      alpha[i] = __expf(m_i[i] - mnew);
      m_i[i] = mnew;
      rs[i] = 0.f;
    }
    _Float16* pw = ldsP[wave];
#pragma unroll
    for (int t = 0; t < 4; ++t) {
#pragma unroll
      for (int i = 0; i < 8; ++i) {
        float p = __expf(st[t][i] - m_i[i]);
        rs[i] += p;
        pw[(i + hi8) * ABN + 16 * t + l16] = (_Float16)p;  // C-layout -> row-major
      }
    }
#pragma unroll
    for (int off = 8; off >= 1; off >>= 1)
#pragma unroll
      for (int i = 0; i < 8; ++i) rs[i] += __shfl_xor(rs[i], off, 32);
#pragma unroll
    for (int i = 0; i < 8; ++i) l_i[i] = l_i[i] * alpha[i] + rs[i];
#pragma unroll
    for (int t = 0; t < 8; ++t)
#pragma unroll
      for (int i = 0; i < 8; ++i) o[t][i] *= alpha[i];

    __syncthreads();  // P stores -> A-fragment reload ordering

    // O += P * V : reload P as A-fragments, V columns from [c][j] tile
    const _Float16* prow = ldsP[wave] + l16 * ABN;
    v16h pf0 = frag_row16(prow, 0, hi8);
    v16h pf1 = frag_row16(prow, 32, hi8);
#pragma unroll
    for (int ct = 0; ct < 8; ++ct) {
      const _Float16* vrow = ldsV + (16 * ct + l16) * ABN;
      o[ct] = __builtin_amdgcn_wmma_f32_16x16x32_f16(false, pf0, false,
                                                     frag_row16(vrow, 0, hi8),
                                                     (short)0, o[ct], false, false);
      o[ct] = __builtin_amdgcn_wmma_f32_16x16x32_f16(false, pf1, false,
                                                     frag_row16(vrow, 32, hi8),
                                                     (short)0, o[ct], false, false);
    }
  }

  // normalize and write ctx as NCHW f32
  float* cb = ctx + (size_t)b * AC * N;
#pragma unroll
  for (int ct = 0; ct < 8; ++ct) {
    int c = 16 * ct + l16;
#pragma unroll
    for (int i = 0; i < 8; ++i) {
      int r = qrow0 + i + hi8;
      cb[(size_t)c * N + r] = o[ct][i] / l_i[i];
    }
  }
}

// ---------------------------------------------------------------------------
// 3x3 conv over channel-concat(ctx, a4) (256 -> 32) + BN + ReLU
// ---------------------------------------------------------------------------
__global__ void k_conv3x3cat_bn_relu(const float* __restrict__ ctx, const float* __restrict__ a4,
                                     const float* __restrict__ w,
                                     const float* __restrict__ g, const float* __restrict__ be,
                                     const float* __restrict__ mn, const float* __restrict__ vr,
                                     float* __restrict__ out, int B, int H, int W) {
  int idx = blockIdx.x * blockDim.x + threadIdx.x;
  int total = B * 32 * H * W;
  if (idx >= total) return;
  int x = idx % W;
  int t = idx / W;
  int y = t % H;
  t /= H;
  int co = t % 32;
  int b = t / 32;
  float acc = 0.f;
  for (int ci = 0; ci < 256; ++ci) {
    const float* src = (ci < 128) ? ctx + (size_t)(b * 128 + ci) * H * W
                                  : a4 + (size_t)(b * 128 + (ci - 128)) * H * W;
    const float* wp = w + ((size_t)co * 256 + ci) * 9;
    for (int ky = 0; ky < 3; ++ky) {
      int yy = y + ky - 1;
      if (yy < 0 || yy >= H) continue;
      for (int kx = 0; kx < 3; ++kx) {
        int xx = x + kx - 1;
        if (xx < 0 || xx >= W) continue;
        acc = fmaf(src[yy * W + xx], wp[ky * 3 + kx], acc);
      }
    }
  }
  float s = bn_scale(g[co], vr[co]);
  float yv = acc * s + (be[co] - mn[co] * s);
  out[idx] = fmaxf(yv, 0.f);
}

// ---------------------------------------------------------------------------
// 1x1 conv (32 -> 19) + bias (classifier head)
// ---------------------------------------------------------------------------
__global__ void k_classer(const float* __restrict__ in, const float* __restrict__ w,
                          const float* __restrict__ bias, float* __restrict__ out,
                          int B, int HW) {
  int idx = blockIdx.x * blockDim.x + threadIdx.x;
  int total = B * 19 * HW;
  if (idx >= total) return;
  int pos = idx % HW;
  int t = idx / HW;
  int co = t % 19;
  int b = t / 19;
  const float* ip = in + (size_t)b * 32 * HW + pos;
  const float* wp = w + co * 32;
  float acc = bias[co];
  for (int ci = 0; ci < 32; ++ci) acc = fmaf(ip[(size_t)ci * HW], wp[ci], acc);
  out[idx] = acc;
}

// ---------------------------------------------------------------------------
extern "C" void kernel_launch(void* const* d_in, const int* in_sizes, int n_in,
                              void* d_out, int out_size, void* d_ws, size_t ws_size,
                              hipStream_t stream) {
  (void)in_sizes; (void)n_in; (void)out_size; (void)ws_size;
  const int B = 2;
  const float* x4 = (const float*)d_in[0];
  const float* x8 = (const float*)d_in[1];
  const float* x16 = (const float*)d_in[2];
  const float* x32 = (const float*)d_in[3];
  auto P = [&](int i) { return (const float*)d_in[i]; };
  // param flattening (dict order): each conv_bn = {w, gamma, beta, mean, var}
  // x4:4..8  x8:9..13  x16:14..18  x32:19..23  conv4:24..28  conv8:29..33
  // q:34..38 k:39..43  v:44..48    convlast:49..53  classer_w:54  classer_b:55

  size_t off = 0;
  auto alloc = [&](size_t bytes) -> void* {
    void* p = (char*)d_ws + off;
    off += (bytes + 255) & ~(size_t)255;
    return p;
  };
  float* a4s = (float*)alloc((size_t)B * 128 * 8192 * 4);
  float* a8s = (float*)alloc((size_t)B * 128 * 2048 * 4);
  float* a16s = (float*)alloc((size_t)B * 128 * 512 * 4);
  float* a32s = (float*)alloc((size_t)B * 128 * 128 * 4);
  float* a32up = (float*)alloc((size_t)B * 128 * 2048 * 4);
  float* a16up = (float*)alloc((size_t)B * 128 * 8192 * 4);
  float* a8c = (float*)alloc((size_t)B * 128 * 2048 * 4);
  float* a8up = (float*)alloc((size_t)B * 128 * 8192 * 4);
  _Float16* qhp = (_Float16*)alloc((size_t)B * 8192 * 128 * 2);
  _Float16* khp = (_Float16*)alloc((size_t)B * 8192 * 128 * 2);
  _Float16* vhp = (_Float16*)alloc((size_t)B * 8192 * 128 * 2);  // channel-major
  float* ctx = (float*)alloc((size_t)B * 128 * 8192 * 4);
  float* cl = (float*)alloc((size_t)B * 32 * 8192 * 4);

  float* a4out = (float*)d_out;                           // tuple[0]: [2,128,64,128]
  float* xfout = (float*)d_out + (size_t)B * 128 * 8192;  // tuple[1]: [2,19,64,128]

  const int T = 256;
  auto g1 = [](int n, int t) { return dim3((unsigned)((n + t - 1) / t)); };

  // stems (1x1 conv + BN + ReLU)
  k_conv1x1_bn_relu_nchw<<<g1(B * 128 * 8192, T), T, 0, stream>>>(x4, P(4), P(5), P(6), P(7), P(8), a4s, B, 64, 128, 8192);
  k_conv1x1_bn_relu_nchw<<<g1(B * 128 * 2048, T), T, 0, stream>>>(x8, P(9), P(10), P(11), P(12), P(13), a8s, B, 128, 128, 2048);
  k_conv1x1_bn_relu_nchw<<<g1(B * 128 * 512, T), T, 0, stream>>>(x16, P(14), P(15), P(16), P(17), P(18), a16s, B, 256, 128, 512);
  k_conv1x1_bn_relu_nchw<<<g1(B * 128 * 128, T), T, 0, stream>>>(x32, P(19), P(20), P(21), P(22), P(23), a32s, B, 1024, 128, 128);

  // upsamples
  k_resize_bilinear<<<g1(B * 128 * 2048, T), T, 0, stream>>>(a32s, a32up, B, 128, 8, 16, 32, 64);
  k_resize_bilinear<<<g1(B * 128 * 8192, T), T, 0, stream>>>(a16s, a16up, B, 128, 16, 32, 64, 128);

  // depthwise 3x3 on fused sums; conv4 result IS tuple output a4
  k_dw3x3_add_bn_relu<<<g1(B * 128 * 8192, T), T, 0, stream>>>(a4s, a16up, P(24), P(25), P(26), P(27), P(28), a4out, B, 128, 64, 128);
  k_dw3x3_add_bn_relu<<<g1(B * 128 * 2048, T), T, 0, stream>>>(a32up, a8s, P(29), P(30), P(31), P(32), P(33), a8c, B, 128, 32, 64);

  k_resize_bilinear<<<g1(B * 128 * 8192, T), T, 0, stream>>>(a8c, a8up, B, 128, 32, 64, 64, 128);

  // projections: q/k token-major f16; v channel-major f16 (TDM-friendly)
  k_conv1x1_bn_relu_tok_h<<<g1(B * 8192 * 128, T), T, 0, stream>>>(a8up, P(34), P(35), P(36), P(37), P(38), qhp, B, 128, 128, 8192);
  k_conv1x1_bn_relu_tok_h<<<g1(B * 8192 * 128, T), T, 0, stream>>>(a4out, P(39), P(40), P(41), P(42), P(43), khp, B, 128, 128, 8192);
  k_conv1x1_bn_relu_nchw_h<<<g1(B * 128 * 8192, T), T, 0, stream>>>(a4out, P(44), P(45), P(46), P(47), P(48), vhp, B, 128, 128, 8192);

  // flash attention (WMMA + TDM): 8192 tokens, head dim 128
  k_flash_attn<<<dim3(8192 / ABM, B), 256, 0, stream>>>(qhp, khp, vhp, ctx, 8192);

  // final convs
  k_conv3x3cat_bn_relu<<<g1(B * 32 * 8192, T), T, 0, stream>>>(ctx, a4out, P(49), P(50), P(51), P(52), P(53), cl, B, 64, 128);
  k_classer<<<g1(B * 19 * 8192, T), T, 0, stream>>>(cl, P(54), P(55), xfout, B, 8192);
}
